// MultiHeadAttention_6219112644808
// MI455X (gfx1250) — matrix operand order repaired and checked by reading, not compiled, not run on hardware
//
#include <hip/hip_runtime.h>
#include <hip/hip_bf16.h>

// ---------- types ----------
typedef __attribute__((ext_vector_type(16))) __bf16        v16bf;
typedef __attribute__((ext_vector_type(8)))  float         v8f;
typedef __attribute__((ext_vector_type(4)))  unsigned int  u32x4;
typedef __attribute__((ext_vector_type(4)))  float         f32x4;
typedef __attribute__((ext_vector_type(4)))  unsigned short u16x4;

union Frag {                 // one WMMA 16-bit operand fragment per lane
    u32x4          q[2];     // two 16-byte loads
    unsigned short u[16];
    v16bf          v;
};

__device__ __forceinline__ unsigned short f32_to_bf16(float f) {
    unsigned int u = __float_as_uint(f);
    unsigned int r = u + 0x7FFFu + ((u >> 16) & 1u);   // round-to-nearest-even
    return (unsigned short)(r >> 16);
}

// A fragment: per ISA 16-bit A 16x32 layout, lane holds two contiguous
// 8-bf16 runs at K = kc+8*half and K = kc+16+8*half.
__device__ __forceinline__ void load_fragA(const unsigned short* __restrict__ row,
                                           int kc, int half, Frag& f) {
    f.q[0] = *(const u32x4*)(row + kc + 8 * half);
    f.q[1] = *(const u32x4*)(row + kc + 16 + 8 * half);
}

// B fragment: lane holds two contiguous 8-bf16 runs at K = kc+8*half and
// K = kc+16+8*half of its column row (weight rows are row-contiguous in K).
__device__ __forceinline__ void load_fragB(const unsigned short* __restrict__ nrow,
                                           int kc, int half, Frag& f) {
    f.q[0] = *(const u32x4*)(nrow + kc + 8 * half);
    f.q[1] = *(const u32x4*)(nrow + kc + 16 + 8 * half);
}

__device__ __forceinline__ void load_rawP(const float* __restrict__ row, int kc,
                                          int half, f32x4 (&r)[4]) {
    r[0] = *(const f32x4*)(row + kc + 8 * half);
    r[1] = *(const f32x4*)(row + kc + 8 * half + 4);
    r[2] = *(const f32x4*)(row + kc + 16 + 8 * half);
    r[3] = *(const f32x4*)(row + kc + 16 + 8 * half + 4);
}

__device__ __forceinline__ void cvt_rawP(const f32x4 (&r)[4], Frag& f) {
    #pragma unroll
    for (int j = 0; j < 4; ++j)
        #pragma unroll
        for (int e = 0; e < 4; ++e)
            f.u[j * 4 + e] = f32_to_bf16(r[j][e]);
}

// ---------- constants ----------
#define BATCH 2
#define SEQ   2048
#define DMODEL 1024
#define NHEADS 16
#define DK    64

// ============================================================
// fp32 -> bf16 bulk convert (n multiple of 4)
// ============================================================
__global__ __launch_bounds__(256) void cvt_bf16(const float* __restrict__ src,
                                                unsigned short* __restrict__ dst,
                                                int n) {
    int i = (blockIdx.x * 256 + threadIdx.x) * 4;
    if (i >= n) return;
    f32x4 f = *(const f32x4*)(src + i);
    u16x4 o;
    o.x = f32_to_bf16(f.x);
    o.y = f32_to_bf16(f.y);
    o.z = f32_to_bf16(f.z);
    o.w = f32_to_bf16(f.w);
    *(u16x4*)(dst + i) = o;
}

// ============================================================
// Generic GEMM: Y = A(MxK) * W^T (W is NxK, torch Linear layout) + bias
// Each wave: 16(M) x 64(N) strip, 4 accumulators, bf16 WMMA f32-acc.
// Manually 2x-unrolled software pipeline with NAMED buffers so every
// fragment stays in VGPRs (no runtime-indexed arrays -> no scratch).
// Requires K % 64 == 0 (true for all call sites: K = 1024).
// mode 0: store bf16, (B,H,S,Dk) layout      (Q/K projections)
// mode 1: store bf16, (B,H,Dk,S) layout      (V projection, transposed)
// mode 2: store fp32, row-major MxN          (output projection)
// ============================================================
__global__ __launch_bounds__(256) void wmma_gemm_bias(
    const unsigned short* __restrict__ A,    // M x K bf16 row-major
    const unsigned short* __restrict__ W,    // N x K bf16 row-major
    const float* __restrict__ bias,          // N fp32
    unsigned short* __restrict__ out_bf,
    float* __restrict__ out_f,
    int M, int N, int K, int mode)
{
    const int lane = threadIdx.x & 31;
    const int wave = threadIdx.x >> 5;
    const int mBase = blockIdx.x * 128 + wave * 16;
    const int nBase = blockIdx.y * 64;
    const int mrow = lane & 15;
    const int half = lane >> 4;
    const int ncol0 = nBase + (lane & 15);

    v8f acc[4];
    #pragma unroll
    for (int t = 0; t < 4; ++t) acc[t] = (v8f){0.f,0.f,0.f,0.f,0.f,0.f,0.f,0.f};

    const unsigned short* arow = A + (long)(mBase + mrow) * K;
    const unsigned short* wrow[4];
    #pragma unroll
    for (int t = 0; t < 4; ++t) wrow[t] = W + (long)(ncol0 + t * 16) * K;

    // pipeline prologue: buffer 0 <- kc = 0
    Frag a0, a1;
    Frag b0[4], b1[4];
    load_fragA(arow, 0, half, a0);
    #pragma unroll
    for (int t = 0; t < 4; ++t) load_fragB(wrow[t], 0, half, b0[t]);

    for (int kc = 0; kc < K; kc += 64) {
        // loads for kc+32 go in flight...
        load_fragA(arow, kc + 32, half, a1);
        #pragma unroll
        for (int t = 0; t < 4; ++t) load_fragB(wrow[t], kc + 32, half, b1[t]);
        __builtin_prefetch(arow + kc + 128, 0, 1);
        // ...while buffer 0 (kc) is consumed
        #pragma unroll
        for (int t = 0; t < 4; ++t)
            acc[t] = __builtin_amdgcn_wmma_f32_16x16x32_bf16(
                false, a0.v, false, b0[t].v, (short)0, acc[t], false, false);
        // loads for kc+64 in flight...
        if (kc + 64 < K) {
            load_fragA(arow, kc + 64, half, a0);
            #pragma unroll
            for (int t = 0; t < 4; ++t) load_fragB(wrow[t], kc + 64, half, b0[t]);
        }
        // ...while buffer 1 (kc+32) is consumed
        #pragma unroll
        for (int t = 0; t < 4; ++t)
            acc[t] = __builtin_amdgcn_wmma_f32_16x16x32_bf16(
                false, a1.v, false, b1[t].v, (short)0, acc[t], false, false);
    }

    // epilogue: C layout -> VGPR r holds (M = r + 8*half, N = lane&15)
    #pragma unroll
    for (int t = 0; t < 4; ++t) {
        const int e = ncol0 + t * 16;
        const float bv = bias ? bias[e] : 0.0f;
        #pragma unroll
        for (int r = 0; r < 8; ++r) {
            const int m = mBase + r + 8 * half;
            const float val = acc[t][r] + bv;
            if (mode == 2) {
                out_f[(long)m * N + e] = val;
            } else {
                const int b = m / SEQ, s = m % SEQ;
                const int h = e >> 6, dk = e & 63;
                long idx;
                if (mode == 0)  // (B,H,S,Dk)
                    idx = (((long)b * NHEADS + h) * SEQ + s) * DK + dk;
                else            // (B,H,Dk,S)
                    idx = (((long)b * NHEADS + h) * DK + dk) * SEQ + s;
                out_bf[idx] = f32_to_bf16(val);
            }
        }
    }
}

// ============================================================
// scores = (Qh . Kh^T) / sqrt(Dk), mask applied, fp32 -> d_out attn region
// K = 64: fully unrolled, all loads issued before the 8 WMMAs.
// grid: x = S/128 m-tiles, y = S/64 n-tiles, z = B*H
// ============================================================
__global__ __launch_bounds__(256) void attn_scores(
    const unsigned short* __restrict__ Qh,   // (B,H,S,Dk) bf16
    const unsigned short* __restrict__ Kh,   // (B,H,S,Dk) bf16
    const int* __restrict__ mask,            // (B,1,1,S)
    float* __restrict__ scores)              // (B,H,S,S)
{
    const int bh = blockIdx.z;
    const int b  = bh >> 4;
    const unsigned short* Q = Qh + (long)bh * SEQ * DK;
    const unsigned short* Kp = Kh + (long)bh * SEQ * DK;
    float* out = scores + (long)bh * SEQ * SEQ;

    const int lane = threadIdx.x & 31;
    const int wave = threadIdx.x >> 5;
    const int mBase = blockIdx.x * 128 + wave * 16;
    const int nBase = blockIdx.y * 64;
    const int mrow = lane & 15;
    const int half = lane >> 4;
    const int ncol0 = nBase + (lane & 15);

    v8f acc[4];
    #pragma unroll
    for (int t = 0; t < 4; ++t) acc[t] = (v8f){0.f,0.f,0.f,0.f,0.f,0.f,0.f,0.f};

    const unsigned short* arow = Q + (long)(mBase + mrow) * DK;

    // issue ALL loads for both K-steps up front
    Frag a0, a1;
    Frag b0[4], b1[4];
    load_fragA(arow, 0, half, a0);
    load_fragA(arow, 32, half, a1);
    #pragma unroll
    for (int t = 0; t < 4; ++t) {
        const unsigned short* krow = Kp + (long)(ncol0 + t * 16) * DK;
        load_fragB(krow, 0,  half, b0[t]);
        load_fragB(krow, 32, half, b1[t]);
    }
    #pragma unroll
    for (int t = 0; t < 4; ++t)
        acc[t] = __builtin_amdgcn_wmma_f32_16x16x32_bf16(
            false, a0.v, false, b0[t].v, (short)0, acc[t], false, false);
    #pragma unroll
    for (int t = 0; t < 4; ++t)
        acc[t] = __builtin_amdgcn_wmma_f32_16x16x32_bf16(
            false, a1.v, false, b1[t].v, (short)0, acc[t], false, false);

    const float scale = 0.125f;  // 1/sqrt(64)
    #pragma unroll
    for (int t = 0; t < 4; ++t) {
        const int n = ncol0 + t * 16;
        const int mk = mask[(long)b * SEQ + n];
        #pragma unroll
        for (int r = 0; r < 8; ++r) {
            const int m = mBase + r + 8 * half;
            float val = acc[t][r] * scale;
            if (mk == 0) val = -1e9f;
            out[(long)m * SEQ + n] = val;
        }
    }
}

// ============================================================
// row softmax in place: one 256-thread block per row of length SEQ
// ============================================================
__global__ __launch_bounds__(256) void softmax_rows(float* __restrict__ p) {
    float* x = p + (long)blockIdx.x * SEQ;
    const int tid = threadIdx.x;
    __shared__ float red[256];

    float v[8];
    float mx = -3.4e38f;
    #pragma unroll
    for (int i = 0; i < 8; ++i) { v[i] = x[tid + i * 256]; mx = fmaxf(mx, v[i]); }
    red[tid] = mx; __syncthreads();
    for (int s = 128; s > 0; s >>= 1) {
        if (tid < s) red[tid] = fmaxf(red[tid], red[tid + s]);
        __syncthreads();
    }
    mx = red[0]; __syncthreads();

    float sum = 0.f;
    #pragma unroll
    for (int i = 0; i < 8; ++i) { v[i] = __expf(v[i] - mx); sum += v[i]; }
    red[tid] = sum; __syncthreads();
    for (int s = 128; s > 0; s >>= 1) {
        if (tid < s) red[tid] += red[tid + s];
        __syncthreads();
    }
    const float inv = 1.0f / red[0];
    #pragma unroll
    for (int i = 0; i < 8; ++i) x[tid + i * 256] = v[i] * inv;
}

// ============================================================
// ctx = P(B,H,S,S fp32) . Vh   with VhT stored (B,H,Dk,S) bf16
// Manually 2x-unrolled pipeline, named buffers only (no scratch).
// Stage order per 64-K step:
//   loads(k+32) -> WMMA(k) -> cvt(k+32) -> loads(k+64) -> WMMA(k+32) -> cvt(k+64)
// output: ctx bf16 row-major (B*S, D), column block h*64..h*64+63
// grid: x = S/128 m-tiles, z = B*H
// ============================================================
__global__ __launch_bounds__(256) void attn_pv(
    const float* __restrict__ P,             // (B,H,S,S) fp32 probs
    const unsigned short* __restrict__ VhT,  // (B,H,Dk,S) bf16
    unsigned short* __restrict__ ctx)        // (B*S, D) bf16
{
    const int bh = blockIdx.z;
    const int b  = bh >> 4;
    const int h  = bh & 15;
    const float* Pr = P + (long)bh * SEQ * SEQ;
    const unsigned short* Vt = VhT + (long)bh * DK * SEQ;

    const int lane = threadIdx.x & 31;
    const int wave = threadIdx.x >> 5;
    const int mBase = blockIdx.x * 128 + wave * 16;
    const int mrow = lane & 15;
    const int half = lane >> 4;
    const int ncol0 = lane & 15;

    v8f acc[4];
    #pragma unroll
    for (int t = 0; t < 4; ++t) acc[t] = (v8f){0.f,0.f,0.f,0.f,0.f,0.f,0.f,0.f};

    const float* arow = Pr + (long)(mBase + mrow) * SEQ;
    const unsigned short* vrow[4];
    #pragma unroll
    for (int t = 0; t < 4; ++t) vrow[t] = Vt + (long)(ncol0 + t * 16) * SEQ;

    f32x4 raw0[4], raw1[4];
    Frag  a0, a1;
    Frag  vb0[4], vb1[4];

    // prologue: buffer 0 <- kc = 0
    load_rawP(arow, 0, half, raw0);
    #pragma unroll
    for (int t = 0; t < 4; ++t) load_fragB(vrow[t], 0, half, vb0[t]);
    cvt_rawP(raw0, a0);

    for (int kc = 0; kc < SEQ; kc += 64) {
        // loads for kc+32 in flight...
        load_rawP(arow, kc + 32, half, raw1);
        #pragma unroll
        for (int t = 0; t < 4; ++t) load_fragB(vrow[t], kc + 32, half, vb1[t]);
        __builtin_prefetch(arow + kc + 128, 0, 1);
        // ...while buffer 0 (kc) is consumed
        #pragma unroll
        for (int t = 0; t < 4; ++t)
            acc[t] = __builtin_amdgcn_wmma_f32_16x16x32_bf16(
                false, a0.v, false, vb0[t].v, (short)0, acc[t], false, false);
        cvt_rawP(raw1, a1);                 // VALU overlaps matrix pipe
        // loads for kc+64 in flight...
        if (kc + 64 < SEQ) {
            load_rawP(arow, kc + 64, half, raw0);
            #pragma unroll
            for (int t = 0; t < 4; ++t) load_fragB(vrow[t], kc + 64, half, vb0[t]);
        }
        // ...while buffer 1 (kc+32) is consumed
        #pragma unroll
        for (int t = 0; t < 4; ++t)
            acc[t] = __builtin_amdgcn_wmma_f32_16x16x32_bf16(
                false, a1.v, false, vb1[t].v, (short)0, acc[t], false, false);
        if (kc + 64 < SEQ) cvt_rawP(raw0, a0);
    }

    #pragma unroll
    for (int t = 0; t < 4; ++t) {
        const int e = h * 64 + ncol0 + t * 16;
        #pragma unroll
        for (int r = 0; r < 8; ++r) {
            const int s = mBase + r + 8 * half;
            ctx[((long)b * SEQ + s) * DMODEL + e] = f32_to_bf16(acc[t][r]);
        }
    }
}

// ============================================================
// host-side launch
// ============================================================
extern "C" void kernel_launch(void* const* d_in, const int* in_sizes, int n_in,
                              void* d_out, int out_size, void* d_ws, size_t ws_size,
                              hipStream_t stream) {
    const float* q    = (const float*)d_in[0];
    const float* k    = (const float*)d_in[1];
    const float* v    = (const float*)d_in[2];
    const int*   mask = (const int*)  d_in[3];
    const float* wq_w = (const float*)d_in[4];
    const float* wq_b = (const float*)d_in[5];
    const float* wk_w = (const float*)d_in[6];
    const float* wk_b = (const float*)d_in[7];
    const float* wv_w = (const float*)d_in[8];
    const float* wv_b = (const float*)d_in[9];
    const float* wo_w = (const float*)d_in[10];
    const float* wo_b = (const float*)d_in[11];

    const long ACT = (long)BATCH * SEQ * DMODEL;      // 4,194,304
    const long WEL = (long)DMODEL * DMODEL;           // 1,048,576
    const int  M   = BATCH * SEQ;                     // 4096

    // workspace layout (ushort counts)
    unsigned short* ws = (unsigned short*)d_ws;
    unsigned short* qbf  = ws;                 // ACT
    unsigned short* kbf  = qbf  + ACT;
    unsigned short* vbf  = kbf  + ACT;
    unsigned short* wqbf = vbf  + ACT;         // WEL each
    unsigned short* wkbf = wqbf + WEL;
    unsigned short* wvbf = wkbf + WEL;
    unsigned short* wobf = wvbf + WEL;
    unsigned short* qh   = wobf + WEL;         // (B,H,S,Dk)
    unsigned short* kh   = qh   + ACT;
    unsigned short* vhT  = kh   + ACT;         // (B,H,Dk,S)
    unsigned short* ctx  = vhT  + ACT;         // (B*S, D)

    float* out_f = (float*)d_out;              // (B,S,D)
    float* attn  = out_f + ACT;                // (B,H,S,S)

    // 1. convert to bf16
    {
        dim3 blk(256);
        cvt_bf16<<<dim3((ACT/4 + 255)/256), blk, 0, stream>>>(q, qbf, (int)ACT);
        cvt_bf16<<<dim3((ACT/4 + 255)/256), blk, 0, stream>>>(k, kbf, (int)ACT);
        cvt_bf16<<<dim3((ACT/4 + 255)/256), blk, 0, stream>>>(v, vbf, (int)ACT);
        cvt_bf16<<<dim3((WEL/4 + 255)/256), blk, 0, stream>>>(wq_w, wqbf, (int)WEL);
        cvt_bf16<<<dim3((WEL/4 + 255)/256), blk, 0, stream>>>(wk_w, wkbf, (int)WEL);
        cvt_bf16<<<dim3((WEL/4 + 255)/256), blk, 0, stream>>>(wv_w, wvbf, (int)WEL);
        cvt_bf16<<<dim3((WEL/4 + 255)/256), blk, 0, stream>>>(wo_w, wobf, (int)WEL);
    }

    // 2. projections (M=4096, N=1024, K=1024), 128x64 tile per block
    {
        dim3 grid(M / 128, DMODEL / 64, 1), blk(256);
        wmma_gemm_bias<<<grid, blk, 0, stream>>>(qbf, wqbf, wq_b, qh,  nullptr,
                                                 M, DMODEL, DMODEL, 0);
        wmma_gemm_bias<<<grid, blk, 0, stream>>>(kbf, wkbf, wk_b, kh,  nullptr,
                                                 M, DMODEL, DMODEL, 0);
        wmma_gemm_bias<<<grid, blk, 0, stream>>>(vbf, wvbf, wv_b, vhT, nullptr,
                                                 M, DMODEL, DMODEL, 1);
    }

    // 3. scores -> attention region of d_out (fp32, masked + scaled)
    {
        dim3 grid(SEQ / 128, SEQ / 64, BATCH * NHEADS), blk(256);
        attn_scores<<<grid, blk, 0, stream>>>(qh, kh, mask, attn);
    }

    // 4. softmax in place (this IS the 2nd output)
    {
        dim3 grid((unsigned)((long)BATCH * NHEADS * SEQ)), blk(256);
        softmax_rows<<<grid, blk, 0, stream>>>(attn);
    }

    // 5. ctx = P . Vh   (bf16 context, row-major B*S x D)
    {
        dim3 grid(SEQ / 128, 1, BATCH * NHEADS), blk(256);
        attn_pv<<<grid, blk, 0, stream>>>(attn, vhT, ctx);
    }

    // 6. output projection -> d_out (fp32)
    {
        dim3 grid(M / 128, DMODEL / 64, 1), blk(256);
        wmma_gemm_bias<<<grid, blk, 0, stream>>>(ctx, wobf, wo_b, nullptr, out_f,
                                                 M, DMODEL, DMODEL, 2);
    }
}
